// CausalSelfAttention_28845000360013
// MI455X (gfx1250) — compile-verified, matrix operand order checked by reading, not run on hardware
//
#include <hip/hip_runtime.h>
#include <hip/hip_bf16.h>

// ---------------------------------------------------------------------------
// CausalSelfAttention for MI455X (gfx1250, wave32, WMMA, bf16 compute).
// Round 2: software-pipelined (register double-buffered) WMMA loops.
// ---------------------------------------------------------------------------

typedef __bf16 bf16_t;
typedef __attribute__((ext_vector_type(8)))  __bf16 bf16x8;
typedef __attribute__((ext_vector_type(16))) __bf16 bf16x16;
typedef __attribute__((ext_vector_type(8)))  float  f32x8;

#define B_  4
#define T_  2048
#define C_  1024
#define H_  16
#define D_  64
#define M_TOT (B_ * T_)          // 8192 rows
#define NEG_INF (-__builtin_inff())

// ---- WMMA wrapper ----------------------------------------------------------
__device__ __forceinline__ f32x8 wmma_bf16(bf16x16 a, bf16x16 b, f32x8 c) {
  return __builtin_amdgcn_wmma_f32_16x16x32_bf16(false, a, false, b,
                                                 (short)0, c, false, false);
}

// ---- fragment loader (16-bit 16x32 A/B layout, ISA 7.12.2) -----------------
// lane row = lane&15; K elements = two contiguous 8-elem chunks selected by
// lane half: [k0 + half*8, +8) and [k0 + 16 + half*8, +8)  -> 2x b128 loads.
__device__ __forceinline__ bf16x16 load_frag(const bf16_t* __restrict__ base,
                                             int ld, int r, int k0, int half) {
  const bf16_t* p = base + (size_t)r * ld + k0 + half * 8;
  bf16x8 lo = *(const bf16x8*)p;
  bf16x8 hi = *(const bf16x8*)(p + 16);
  union { bf16x16 v; bf16x8 h[2]; } u;
  u.h[0] = lo; u.h[1] = hi;
  return u.v;
}

// ---- wave32 16-lane-group reductions ---------------------------------------
__device__ __forceinline__ float rowmax16(float v) {
#pragma unroll
  for (int m = 1; m < 16; m <<= 1) v = fmaxf(v, __shfl_xor(v, m, 32));
  return v;
}
__device__ __forceinline__ float rowsum16(float v) {
#pragma unroll
  for (int m = 1; m < 16; m <<= 1) v += __shfl_xor(v, m, 32);
  return v;
}

// ---------------------------------------------------------------------------
// Shared pipelined 32x64 GEMM tile over K = C_ (A row-major, B transposed).
// ---------------------------------------------------------------------------
struct FragSet {
  bf16x16 a[2];
  bf16x16 b[4];
};

__device__ __forceinline__ void load_set(FragSet& f,
                                         const bf16_t* __restrict__ A,
                                         const bf16_t* __restrict__ BT,
                                         int m0, int n0, int k0,
                                         int col, int half) {
  f.a[0] = load_frag(A, C_, m0 + col, k0, half);
  f.a[1] = load_frag(A, C_, m0 + 16 + col, k0, half);
#pragma unroll
  for (int j = 0; j < 4; ++j)
    f.b[j] = load_frag(BT, C_, n0 + j * 16 + col, k0, half);
}

__device__ __forceinline__ void mma_set(const FragSet& f, f32x8 acc[2][4]) {
#pragma unroll
  for (int j = 0; j < 4; ++j) {
    acc[0][j] = wmma_bf16(f.a[0], f.b[j], acc[0][j]);
    acc[1][j] = wmma_bf16(f.a[1], f.b[j], acc[1][j]);
  }
}

// Double-buffered: loads for the next 32-K half-step are in flight while the
// current half-step's 8 WMMAs execute.
__device__ __forceinline__ void gemm_tile_32x64(const bf16_t* __restrict__ A,
                                                const bf16_t* __restrict__ BT,
                                                int m0, int n0, int col,
                                                int half, f32x8 acc[2][4]) {
  FragSet f0, f1;
  load_set(f0, A, BT, m0, n0, 0, col, half);
#pragma unroll
  for (int k0 = 0; k0 < C_; k0 += 64) {
    load_set(f1, A, BT, m0, n0, k0 + 32, col, half);   // prefetch half-step
    mma_set(f0, acc);
    if (k0 + 64 < C_) load_set(f0, A, BT, m0, n0, k0 + 64, col, half);
    mma_set(f1, acc);
  }
}

// ---------------------------------------------------------------------------
// Precision conversion kernels
// ---------------------------------------------------------------------------
__global__ void f32_to_bf16_vec8_kernel(const float* __restrict__ in,
                                        bf16_t* __restrict__ out, int n8) {
  int i = blockIdx.x * 256 + threadIdx.x;
  if (i < n8) {
    const float4* p = (const float4*)in + (size_t)i * 2;
    float4 x0 = p[0], x1 = p[1];
    union { bf16x8 v; bf16_t e[8]; } u;
    u.e[0] = (bf16_t)x0.x; u.e[1] = (bf16_t)x0.y;
    u.e[2] = (bf16_t)x0.z; u.e[3] = (bf16_t)x0.w;
    u.e[4] = (bf16_t)x1.x; u.e[5] = (bf16_t)x1.y;
    u.e[6] = (bf16_t)x1.z; u.e[7] = (bf16_t)x1.w;
    *((bf16x8*)out + i) = u.v;
  }
}

// in[rows][cols] (f32) -> out[cols][rows] (bf16)
__global__ void transpose_f32_to_bf16_kernel(const float* __restrict__ in,
                                             bf16_t* __restrict__ out,
                                             int rows, int cols) {
  int i = blockIdx.x * 256 + threadIdx.x;
  if (i < rows * cols) {
    int r = i / cols, c = i % cols;
    out[(size_t)c * rows + r] = (bf16_t)in[i];
  }
}

// ---------------------------------------------------------------------------
// QKV GEMM: [8192,1024] x [1024,3072] + bias -> Q/K [B,H,T,D], V^T [B,H,D,T]
// ---------------------------------------------------------------------------
__global__ __launch_bounds__(256) void qkv_gemm_kernel(
    const bf16_t* __restrict__ xb,   // [8192,1024]
    const bf16_t* __restrict__ waT,  // [3072,1024]
    const float*  __restrict__ bias, // [3072]
    bf16_t* __restrict__ qb, bf16_t* __restrict__ kb,
    bf16_t* __restrict__ vT) {
  const int lane = threadIdx.x & 31, wv = threadIdx.x >> 5;
  const int half = lane >> 4, col = lane & 15;
  int w  = blockIdx.x * 8 + wv;      // 12288 waves
  int mt = w & 255;                  // 256 M-tiles (32 rows)
  int nt = w >> 8;                   // 48 N-quads (64 cols)
  int m0 = mt * 32, n0 = nt * 64;

  f32x8 acc[2][4] = {};
  gemm_tile_32x64(xb, waT, m0, n0, col, half, acc);

#pragma unroll
  for (int j = 0; j < 4; ++j) {
    int n = n0 + j * 16 + col;
    float bv = bias[n];
    int which = n >> 10, c = n & (C_ - 1);
    int h = c >> 6, d = c & (D_ - 1);
#pragma unroll
    for (int mi = 0; mi < 2; ++mi) {
      int gmb = m0 + mi * 16 + 8 * half;       // first of 8 consecutive rows
      int bidx = gmb >> 11;
      size_t bh = (size_t)(bidx * H_ + h);
      if (which == 2) {
        // lane holds 8 consecutive t for fixed d: one packed b128 store
        union { bf16x8 v; bf16_t e[8]; } pk;
#pragma unroll
        for (int rr = 0; rr < 8; ++rr)
          pk.e[rr] = (bf16_t)(acc[mi][j][rr] + bv);
        *(bf16x8*)&vT[(bh * D_ + d) * T_ + (gmb & (T_ - 1))] = pk.v;
      } else {
        bf16_t* dst = (which == 0) ? qb : kb;
#pragma unroll
        for (int rr = 0; rr < 8; ++rr) {
          int t = (gmb + rr) & (T_ - 1);
          dst[(bh * T_ + t) * D_ + d] = (bf16_t)(acc[mi][j][rr] + bv);
        }
      }
    }
  }
}

// ---------------------------------------------------------------------------
// Flash attention: one wave per (b,h, 16-query tile), pipelined 32-key blocks.
// ---------------------------------------------------------------------------
__device__ __forceinline__ void load_kblock(bf16x16 kf[4],
                                            const bf16_t* __restrict__ K,
                                            int k0, int col, int half) {
  kf[0] = load_frag(K, D_, k0 + col, 0, half);
  kf[1] = load_frag(K, D_, k0 + col, 32, half);
  kf[2] = load_frag(K, D_, k0 + 16 + col, 0, half);
  kf[3] = load_frag(K, D_, k0 + 16 + col, 32, half);
}

__global__ __launch_bounds__(128) void attn_kernel(
    const bf16_t* __restrict__ qb, const bf16_t* __restrict__ kb,
    const bf16_t* __restrict__ vT, bf16_t* __restrict__ yb) {
  __shared__ bf16_t plds[4][16][32];   // per-wave P tile (D-layout -> A-layout)

  const int lane = threadIdx.x & 31, wv = threadIdx.x >> 5;
  const int half = lane >> 4, col = lane & 15;
  int w  = blockIdx.x * 4 + wv;        // 8192 waves
  int qt = w & 127, bh = w >> 7;
  int q0 = qt * 16;

  const bf16_t* Q = qb + (size_t)bh * T_ * D_;
  const bf16_t* K = kb + (size_t)bh * T_ * D_;
  const bf16_t* V = vT + (size_t)bh * D_ * T_;

  bf16x16 qa0 = load_frag(Q, D_, q0 + col, 0, half);
  bf16x16 qa1 = load_frag(Q, D_, q0 + col, 32, half);

  f32x8 o[4] = {};
  float mrun[8], lrun[8];
#pragma unroll
  for (int rr = 0; rr < 8; ++rr) { mrun[rr] = NEG_INF; lrun[rr] = 0.f; }

  const float scale = 0.125f;                 // 1/sqrt(64)
  int nblk = (q0 + 47) >> 5;                  // causal: k0 <= q0+15

  bf16x16 kf[4], nf[4];
  load_kblock(kf, K, 0, col, half);

  for (int kblk = 0; kblk < nblk; ++kblk) {
    int k0 = kblk * 32;

    f32x8 s0 = {}, s1 = {};
    s0 = wmma_bf16(qa0, kf[0], s0);
    s0 = wmma_bf16(qa1, kf[1], s0);
    s1 = wmma_bf16(qa0, kf[2], s1);
    s1 = wmma_bf16(qa1, kf[3], s1);

    // Issue independent loads now; softmax VALU below covers their latency.
    bf16x16 vf[4];
#pragma unroll
    for (int j = 0; j < 4; ++j)
      vf[j] = load_frag(V, T_, j * 16 + col, k0, half);
    bool more = (kblk + 1 < nblk);
    if (more) load_kblock(nf, K, k0 + 32, col, half);

#pragma unroll
    for (int rr = 0; rr < 8; ++rr) {
      int qg = q0 + rr + 8 * half;
      float v0 = (k0 + col      <= qg) ? s0[rr] * scale : NEG_INF;
      float v1 = (k0 + 16 + col <= qg) ? s1[rr] * scale : NEG_INF;
      float bm   = rowmax16(fmaxf(v0, v1));
      float mnew = fmaxf(mrun[rr], bm);
      float alpha = __expf(mrun[rr] - mnew);
      float p0 = __expf(v0 - mnew);
      float p1 = __expf(v1 - mnew);
      lrun[rr] = lrun[rr] * alpha + rowsum16(p0 + p1);
      mrun[rr] = mnew;
      int mr = rr + 8 * half;
      plds[wv][mr][col]      = (bf16_t)p0;
      plds[wv][mr][16 + col] = (bf16_t)p1;
#pragma unroll
      for (int j = 0; j < 4; ++j) o[j][rr] *= alpha;
    }

    // Re-read P in A-fragment layout (same-wave DS ops are in-order; the
    // compiler inserts the s_wait_dscnt for the RAW dependency).
    bf16x8 plo = *(const bf16x8*)&plds[wv][col][half * 8];
    bf16x8 phi = *(const bf16x8*)&plds[wv][col][16 + half * 8];
    union { bf16x16 v; bf16x8 h[2]; } pu;
    pu.h[0] = plo; pu.h[1] = phi;

#pragma unroll
    for (int j = 0; j < 4; ++j)        // O[16x64] += P[16x32] @ V[32x64]
      o[j] = wmma_bf16(pu.v, vf[j], o[j]);

    if (more) {
#pragma unroll
      for (int j = 0; j < 4; ++j) kf[j] = nf[j];
    }
  }

  int bidx = bh >> 4, h = bh & (H_ - 1);
#pragma unroll
  for (int rr = 0; rr < 8; ++rr) {
    float inv = 1.f / lrun[rr];
    int t = q0 + rr + 8 * half;
    size_t rowoff = ((size_t)(bidx * T_ + t)) * C_ + h * D_;
#pragma unroll
    for (int j = 0; j < 4; ++j)
      yb[rowoff + j * 16 + col] = (bf16_t)(o[j][rr] * inv);
  }
}

// ---------------------------------------------------------------------------
// Output projection: [8192,1024] x [1024,1024] + bias -> fp32 out.
// ---------------------------------------------------------------------------
__global__ __launch_bounds__(256) void proj_gemm_kernel(
    const bf16_t* __restrict__ yb, const bf16_t* __restrict__ wpT,
    const float* __restrict__ bias, float* __restrict__ out) {
  const int lane = threadIdx.x & 31, wv = threadIdx.x >> 5;
  const int half = lane >> 4, col = lane & 15;
  int w  = blockIdx.x * 8 + wv;      // 4096 waves
  int mt = w & 255;
  int nt = w >> 8;                   // 16 N-quads
  int m0 = mt * 32, n0 = nt * 64;

  f32x8 acc[2][4] = {};
  gemm_tile_32x64(yb, wpT, m0, n0, col, half, acc);

#pragma unroll
  for (int j = 0; j < 4; ++j) {
    int n = n0 + j * 16 + col;
    float bv = bias[n];
#pragma unroll
    for (int mi = 0; mi < 2; ++mi) {
#pragma unroll
      for (int rr = 0; rr < 8; ++rr) {
        int gm = m0 + mi * 16 + rr + 8 * half;
        out[(size_t)gm * C_ + n] = acc[mi][j][rr] + bv;
      }
    }
  }
}

// ---------------------------------------------------------------------------
extern "C" void kernel_launch(void* const* d_in, const int* in_sizes, int n_in,
                              void* d_out, int out_size, void* d_ws,
                              size_t ws_size, hipStream_t stream) {
  const float* x      = (const float*)d_in[0];
  const float* w_attn = (const float*)d_in[1];
  const float* b_attn = (const float*)d_in[2];
  const float* w_proj = (const float*)d_in[3];
  const float* b_proj = (const float*)d_in[4];
  float* out = (float*)d_out;

  char* ws = (char*)d_ws;
  size_t off = 0;
  bf16_t* xb  = (bf16_t*)(ws + off); off += (size_t)M_TOT * C_ * 2;
  bf16_t* waT = (bf16_t*)(ws + off); off += (size_t)3 * C_ * C_ * 2;
  bf16_t* wpT = (bf16_t*)(ws + off); off += (size_t)C_ * C_ * 2;
  bf16_t* qb  = (bf16_t*)(ws + off); off += (size_t)M_TOT * C_ * 2;
  bf16_t* kb  = (bf16_t*)(ws + off); off += (size_t)M_TOT * C_ * 2;
  bf16_t* vT  = (bf16_t*)(ws + off); off += (size_t)M_TOT * C_ * 2;
  bf16_t* yb  = (bf16_t*)(ws + off); off += (size_t)M_TOT * C_ * 2;

  int n8 = (M_TOT * C_) / 8;   // 1M vec8 elements
  f32_to_bf16_vec8_kernel<<<n8 / 256, 256, 0, stream>>>(x, xb, n8);
  int nwa = C_ * 3 * C_;
  transpose_f32_to_bf16_kernel<<<nwa / 256, 256, 0, stream>>>(w_attn, waT,
                                                              C_, 3 * C_);
  int nwp = C_ * C_;
  transpose_f32_to_bf16_kernel<<<nwp / 256, 256, 0, stream>>>(w_proj, wpT,
                                                              C_, C_);

  // QKV: 256 M-tiles * 48 N-quads = 12288 waves / 8 per block
  qkv_gemm_kernel<<<1536, 256, 0, stream>>>(xb, waT, b_attn, qb, kb, vT);

  // Attention: 64 (b,h) * 128 query tiles = 8192 waves / 4 per block
  attn_kernel<<<2048, 128, 0, stream>>>(qb, kb, vT, yb);

  // Proj: 256 * 16 = 4096 waves / 8 per block
  proj_gemm_kernel<<<512, 256, 0, stream>>>(yb, wpT, b_proj, out);
}
